// MAGNODecoder_72816875536551
// MI455X (gfx1250) — matrix operand order repaired and checked by reading, not compiled, not run on hardware
//
#include <hip/hip_runtime.h>
#include <hip/hip_fp16.h>
#include <math.h>

typedef __attribute__((ext_vector_type(16))) _Float16 v16h;
typedef __attribute__((ext_vector_type(8)))  _Float16 v8h;
typedef __attribute__((ext_vector_type(2)))  _Float16 v2h;
typedef __attribute__((ext_vector_type(8)))  float    v8f;

#define NLAT  2048
#define NQTOT 16384
#define BATCH 2
#define KNN   32
#define HDIM  64
#define PHDIM 256
#define CINCH 64
#define TOTQ  (BATCH * NQTOT)

// branch-free GELU (tanh form); 1 - 2/(e+1) is well-behaved at +-inf
__device__ __forceinline__ float gelu_fast(float x) {
    float u = x * (0.7978845608028654f + 0.035677408136300125f * x * x);
    float e = __expf(2.0f * u);
    float t = 1.0f - 2.0f / (e + 1.0f);
    return 0.5f * x * (1.0f + t);
}

// ---------------------------------------------------------------------------
// prep: convert f32 [K][N] weight matrices into fp16 WMMA B-fragment layout.
// B layout (16-bit, 32K x 16N per fragment): lanes 0-15 carry N=0..15 with
// K {0..7,16..23}, lanes 16-31 the same N with K {8..15,24..31}; per-lane contiguous 16 halves.
// ---------------------------------------------------------------------------
__device__ __forceinline__ void frag_convert(const float* __restrict__ W,
                                             _Float16* __restrict__ dst,
                                             int KK, int NN, int tid, int nth) {
    int total = KK * NN;
    int ntiles = NN >> 4;
    for (int i = tid; i < total; i += nth) {
        int h    = i & 15;
        int lane = (i >> 4) & 31;
        int frag = i >> 9;
        int kc = frag / ntiles, nt = frag % ntiles;
        int n = nt * 16 + (lane & 15);
        int k = kc * 32 + (h & 7) + ((h >= 8) ? 16 : 0) + ((lane >= 16) ? 8 : 0);
        dst[i] = (_Float16)W[k * NN + n];
    }
}

__global__ void prep_kernel(const float* __restrict__ w_e1,
                            const float* __restrict__ w_e2,
                            const float* __restrict__ w_p0,
                            _Float16* __restrict__ fB1,
                            _Float16* __restrict__ fB2,
                            _Float16* __restrict__ fP0) {
    int tid = blockIdx.x * blockDim.x + threadIdx.x;
    int nth = gridDim.x * blockDim.x;
    frag_convert(w_e1, fB1, HDIM, HDIM, tid, nth);
    frag_convert(w_e2, fB2, HDIM, HDIM, tid, nth);
    frag_convert(w_p0, fP0, CINCH, PHDIM, tid, nth);
}

// ---------------------------------------------------------------------------
// top-K: one wave per query; latent coords + per-wave d2 slab in LDS,
// 32 iterative min-extractions with wave-wide shfl_xor reductions.
// ---------------------------------------------------------------------------
#define TKWAVES 4
__global__ __launch_bounds__(128) void topk_kernel(
        const float* __restrict__ ycoord, const float* __restrict__ qcoord,
        int* __restrict__ outIdx, float* __restrict__ outD2) {
    __shared__ float sY[NLAT * 2];
    __shared__ float sD2[TKWAVES][NLAT];
    int tid = threadIdx.x;
    for (int i = tid; i < NLAT * 2; i += 128) sY[i] = ycoord[i];
    __syncthreads();

    int wave = tid >> 5, lane = tid & 31;
    int g = blockIdx.x * TKWAVES + wave;
    float qx = qcoord[g * 2 + 0], qy = qcoord[g * 2 + 1];
    float* d2 = sD2[wave];

    for (int t = 0; t < NLAT / 32; ++t) {
        int j = t * 32 + lane;
        float dx = qx - sY[2 * j], dy = qy - sY[2 * j + 1];
        d2[j] = dx * dx + dy * dy;
    }
    asm volatile("s_wait_dscnt 0" ::: "memory");

    float myD2 = 0.f; int myIdx = 0;
    for (int k = 0; k < KNN; ++k) {
        float best = 3.4e38f; int bi = NLAT;
        for (int t = 0; t < NLAT / 32; ++t) {
            int j = lane + 32 * t;
            float v = d2[j];
            bool better = (v < best) || (v == best && j < bi);
            best = better ? v : best;
            bi   = better ? j : bi;
        }
        #pragma unroll
        for (int off = 16; off > 0; off >>= 1) {
            float ov = __shfl_xor(best, off, 32);
            int   oi = __shfl_xor(bi, off, 32);
            bool better = (ov < best) || (ov == best && oi < bi);
            best = better ? ov : best;
            bi   = better ? oi : bi;
        }
        if (lane == k) { myD2 = best; myIdx = bi; }
        if (lane == (bi & 31)) d2[bi] = 3.4e38f;
        asm volatile("s_wait_dscnt 0" ::: "memory");
    }
    outIdx[(size_t)g * KNN + lane] = myIdx;
    outD2[(size_t)g * KNN + lane]  = myD2;
}

// ---------------------------------------------------------------------------
// WMMA fragment loads
// A layout (16-bit 16x32): lanes 0-15 row M=L, K {0..7,16..23};
// lanes 16-31 row M=L-16, K {8..15,24..31}.
// ---------------------------------------------------------------------------
__device__ __forceinline__ v16h load_afrag_h16(const _Float16* __restrict__ hbuf,
                                               int lane, int mt, int kc) {
    int row  = mt * 16 + (lane & 15);
    int base = kc * 32 + ((lane < 16) ? 0 : 8);
    const _Float16* p = hbuf + row * HDIM + base;
    v8h lo = *(const v8h*)(p);
    v8h hi = *(const v8h*)(p + 16);
    v16h a;
    #pragma unroll
    for (int i = 0; i < 8; ++i) { a[i] = lo[i]; a[i + 8] = hi[i]; }
    return a;
}

__device__ __forceinline__ v16h load_bfrag(const _Float16* __restrict__ frag,
                                           int lane, int kc, int nt, int ntiles) {
    return *(const v16h*)(frag + ((((kc * ntiles + nt) * 32) + lane) << 4));
}

// ---------------------------------------------------------------------------
// edge MLP: one wave per query; 8 waves/block amortize the 16KB fragment
// staging. Scales share neighbors + MLP; combined per-row weight
// w_k = sum_s sw_s * valid_s[k]/cnt_s is applied once in the reduction.
// ---------------------------------------------------------------------------
#define EWAVES 8
__global__ __launch_bounds__(256) void edge_kernel(
        const float* __restrict__ ycoord, const float* __restrict__ rndata,
        const float* __restrict__ qcoord,
        const float* __restrict__ w_e0, const float* __restrict__ b_e0,
        const float* __restrict__ b_e1, const float* __restrict__ b_e2,
        const float* __restrict__ w_sw0, const float* __restrict__ b_sw0,
        const float* __restrict__ w_sw1, const float* __restrict__ b_sw1,
        const _Float16* __restrict__ fB1, const _Float16* __restrict__ fB2,
        const int* __restrict__ topIdx, const float* __restrict__ topD2,
        float* __restrict__ decoded) {
    __shared__ __align__(32) _Float16 sB1f[HDIM * HDIM];
    __shared__ __align__(32) _Float16 sB2f[HDIM * HDIM];
    __shared__ __align__(16) _Float16 sH[EWAVES][KNN * HDIM];
    __shared__ float sW0[4 * HDIM];
    __shared__ float sBias[3 * HDIM];
    __shared__ int   sIdx[EWAVES][KNN];
    __shared__ float sWk[EWAVES][KNN];

    int tid = threadIdx.x;
    {   // cooperative staging, b128-wide
        const uint4* s1 = (const uint4*)fB1;
        const uint4* s2 = (const uint4*)fB2;
        uint4* d1 = (uint4*)sB1f;
        uint4* d2v = (uint4*)sB2f;
        #pragma unroll
        for (int i = tid; i < HDIM * HDIM * 2 / 16; i += 256) { d1[i] = s1[i]; d2v[i] = s2[i]; }
        sW0[tid] = w_e0[tid];
        if (tid < 64) { sBias[tid] = b_e0[tid]; sBias[64 + tid] = b_e1[tid]; sBias[128 + tid] = b_e2[tid]; }
    }
    __syncthreads();

    int wave = tid >> 5, lane = tid & 31;
    int g = blockIdx.x * EWAVES + wave;            // query id
    int b = g / NQTOT;

    int   nb = topIdx[(size_t)g * KNN + lane];
    float d2 = topD2[(size_t)g * KNN + lane];
    float yx = ycoord[nb * 2 + 0], yy = ycoord[nb * 2 + 1];
    float qx = qcoord[g * 2 + 0],  qy = qcoord[g * 2 + 1];

    // scale-mix gate: Linear(2->16) -> ReLU -> Linear(16->2) -> softmax
    float s0 = b_sw1[0], s1 = b_sw1[1];
    #pragma unroll
    for (int i = 0; i < 16; ++i) {
        float hsw = fmaxf(qx * w_sw0[i] + qy * w_sw0[16 + i] + b_sw0[i], 0.f);
        s0 += hsw * w_sw1[i * 2 + 0];
        s1 += hsw * w_sw1[i * 2 + 1];
    }
    float mx = fmaxf(s0, s1);
    float e0 = __expf(s0 - mx), e1 = __expf(s1 - mx);
    float inv = 1.f / (e0 + e1);
    float swA = e0 * inv, swB = e1 * inv;

    const float R1 = 0.055f * 0.055f, R2 = 0.11f * 0.11f;
    bool v1 = d2 <= R1, v2 = d2 <= R2;
    float c1 = fmaxf((float)__popcll(__ballot(v1)), 1.f);
    float c2 = fmaxf((float)__popcll(__ballot(v2)), 1.f);
    sIdx[wave][lane] = nb;
    sWk[wave][lane]  = swA * (v1 ? 1.f : 0.f) / c1 + swB * (v2 ? 1.f : 0.f) / c2;

    // layer 0 (K=4, VALU): feats = {yx,yy,qx,qy} @ w_e0 + b_e0 -> GELU -> LDS fp16
    _Float16* hbuf = sH[wave];
    #pragma unroll 8
    for (int n = 0; n < HDIM; n += 2) {
        float a0 = sBias[n]     + yx * sW0[n]     + yy * sW0[64 + n]
                 + qx * sW0[128 + n]     + qy * sW0[192 + n];
        float a1 = sBias[n + 1] + yx * sW0[n + 1] + yy * sW0[64 + n + 1]
                 + qx * sW0[128 + n + 1] + qy * sW0[192 + n + 1];
        v2h p; p[0] = (_Float16)gelu_fast(a0); p[1] = (_Float16)gelu_fast(a1);
        *(v2h*)(hbuf + lane * HDIM + n) = p;
    }
    asm volatile("s_wait_dscnt 0" ::: "memory");

    v8f acc[2][4];
    // ---- layer 1: [32x64] @ [64x64] via 16 WMMAs, bias preloaded in C ----
    #pragma unroll
    for (int mt = 0; mt < 2; ++mt)
        #pragma unroll
        for (int nt = 0; nt < 4; ++nt) {
            float bv = sBias[64 + nt * 16 + (lane & 15)];
            v8f c;
            #pragma unroll
            for (int i = 0; i < 8; ++i) c[i] = bv;
            acc[mt][nt] = c;
        }
    #pragma unroll
    for (int kc = 0; kc < 2; ++kc) {
        v16h a0 = load_afrag_h16(hbuf, lane, 0, kc);
        v16h a1 = load_afrag_h16(hbuf, lane, 1, kc);
        #pragma unroll
        for (int nt = 0; nt < 4; ++nt) {
            v16h bf = load_bfrag(sB1f, lane, kc, nt, 4);
            acc[0][nt] = __builtin_amdgcn_wmma_f32_16x16x32_f16(false, a0, false, bf, (short)0, acc[0][nt], false, false);
            acc[1][nt] = __builtin_amdgcn_wmma_f32_16x16x32_f16(false, a1, false, bf, (short)0, acc[1][nt], false, false);
        }
    }
    // GELU + store back to activation tile (C layout: row = mt*16 + j + half*8)
    #pragma unroll
    for (int mt = 0; mt < 2; ++mt)
        #pragma unroll
        for (int nt = 0; nt < 4; ++nt) {
            int rbase = mt * 16 + ((lane < 16) ? 0 : 8);
            int n = nt * 16 + (lane & 15);
            #pragma unroll
            for (int j = 0; j < 8; ++j)
                hbuf[(rbase + j) * HDIM + n] = (_Float16)gelu_fast(acc[mt][nt][j]);
        }
    asm volatile("s_wait_dscnt 0" ::: "memory");

    // ---- layer 2: [32x64] @ [64x64], no activation ----
    #pragma unroll
    for (int mt = 0; mt < 2; ++mt)
        #pragma unroll
        for (int nt = 0; nt < 4; ++nt) {
            float bv = sBias[128 + nt * 16 + (lane & 15)];
            v8f c;
            #pragma unroll
            for (int i = 0; i < 8; ++i) c[i] = bv;
            acc[mt][nt] = c;
        }
    #pragma unroll
    for (int kc = 0; kc < 2; ++kc) {
        v16h a0 = load_afrag_h16(hbuf, lane, 0, kc);
        v16h a1 = load_afrag_h16(hbuf, lane, 1, kc);
        #pragma unroll
        for (int nt = 0; nt < 4; ++nt) {
            v16h bf = load_bfrag(sB2f, lane, kc, nt, 4);
            acc[0][nt] = __builtin_amdgcn_wmma_f32_16x16x32_f16(false, a0, false, bf, (short)0, acc[0][nt], false, false);
            acc[1][nt] = __builtin_amdgcn_wmma_f32_16x16x32_f16(false, a1, false, bf, (short)0, acc[1][nt], false, false);
        }
    }

    // ---- masked, scale-mixed mean: dec[c] = sum_k h3[k][c]*f[k][c]*w_k ----
    float dec0 = 0.f, dec1 = 0.f, dec2 = 0.f, dec3 = 0.f;
    const float* fbase = rndata + (size_t)b * NLAT * CINCH;
    #pragma unroll
    for (int mt = 0; mt < 2; ++mt)
        #pragma unroll
        for (int j = 0; j < 8; ++j) {
            int row = mt * 16 + j + ((lane < 16) ? 0 : 8);
            float wkr = sWk[wave][row];
            const float* fr = fbase + (size_t)sIdx[wave][row] * CINCH + (lane & 15);
            dec0 += acc[mt][0][j] * fr[0]  * wkr;
            dec1 += acc[mt][1][j] * fr[16] * wkr;
            dec2 += acc[mt][2][j] * fr[32] * wkr;
            dec3 += acc[mt][3][j] * fr[48] * wkr;
        }
    dec0 += __shfl_xor(dec0, 16, 32);
    dec1 += __shfl_xor(dec1, 16, 32);
    dec2 += __shfl_xor(dec2, 16, 32);
    dec3 += __shfl_xor(dec3, 16, 32);
    if (lane < 16) {
        float* dq = decoded + (size_t)g * CINCH;
        dq[lane] = dec0; dq[16 + lane] = dec1; dq[32 + lane] = dec2; dq[48 + lane] = dec3;
    }
}

// ---------------------------------------------------------------------------
// projection MLP: GELU(decoded @ w_p0[64x256]) @ w_p1[256x3] + b.
// 8 waves/block amortize the 32KB fragment staging; one wave per 16-row tile.
// ---------------------------------------------------------------------------
#define PWAVES 8
__global__ __launch_bounds__(256) void proj_kernel(
        const float* __restrict__ decoded, const _Float16* __restrict__ fP0,
        const float* __restrict__ b_p0, const float* __restrict__ w_p1,
        const float* __restrict__ b_p1, float* __restrict__ out) {
    __shared__ __align__(32) _Float16 sP0f[CINCH * PHDIM];
    __shared__ float sBp0[PHDIM];
    __shared__ float sWp1[PHDIM * 3];

    int tid = threadIdx.x;
    {
        const uint4* s = (const uint4*)fP0;
        uint4* d = (uint4*)sP0f;
        #pragma unroll
        for (int i = tid; i < CINCH * PHDIM * 2 / 16; i += 256) d[i] = s[i];
        sBp0[tid] = b_p0[tid];
        for (int i = tid; i < PHDIM * 3; i += 256) sWp1[i] = w_p1[i];
    }
    __syncthreads();

    int wave = tid >> 5, lane = tid & 31;
    int t = blockIdx.x * PWAVES + wave;            // tile id; rows t*16..t*16+15
    int row = t * 16 + (lane & 15);

    v16h aA[2];
    #pragma unroll
    for (int kc = 0; kc < 2; ++kc) {
        const float* dr = decoded + (size_t)row * CINCH + kc * 32 + ((lane < 16) ? 0 : 8);
        v16h a;
        #pragma unroll
        for (int i = 0; i < 8; ++i) { a[i] = (_Float16)dr[i]; a[i + 8] = (_Float16)dr[16 + i]; }
        aA[kc] = a;
    }

    float po[8][3];
    #pragma unroll
    for (int j = 0; j < 8; ++j) { po[j][0] = 0.f; po[j][1] = 0.f; po[j][2] = 0.f; }

    for (int nt = 0; nt < PHDIM / 16; ++nt) {
        int n = nt * 16 + (lane & 15);
        float bv = sBp0[n];
        v8f c;
        #pragma unroll
        for (int i = 0; i < 8; ++i) c[i] = bv;
        c = __builtin_amdgcn_wmma_f32_16x16x32_f16(false, aA[0], false,
                load_bfrag(sP0f, lane, 0, nt, PHDIM / 16), (short)0, c, false, false);
        c = __builtin_amdgcn_wmma_f32_16x16x32_f16(false, aA[1], false,
                load_bfrag(sP0f, lane, 1, nt, PHDIM / 16), (short)0, c, false, false);
        float w0 = sWp1[n * 3 + 0], w1 = sWp1[n * 3 + 1], w2 = sWp1[n * 3 + 2];
        #pragma unroll
        for (int j = 0; j < 8; ++j) {
            float h = gelu_fast(c[j]);
            po[j][0] += h * w0; po[j][1] += h * w1; po[j][2] += h * w2;
        }
    }
    #pragma unroll
    for (int off = 1; off < 16; off <<= 1)
        #pragma unroll
        for (int j = 0; j < 8; ++j) {
            po[j][0] += __shfl_xor(po[j][0], off, 32);
            po[j][1] += __shfl_xor(po[j][1], off, 32);
            po[j][2] += __shfl_xor(po[j][2], off, 32);
        }
    if ((lane & 15) == 0) {
        int r0 = t * 16 + ((lane < 16) ? 0 : 8);
        float bo0 = b_p1[0], bo1 = b_p1[1], bo2 = b_p1[2];
        #pragma unroll
        for (int j = 0; j < 8; ++j) {
            out[(size_t)(r0 + j) * 3 + 0] = po[j][0] + bo0;
            out[(size_t)(r0 + j) * 3 + 1] = po[j][1] + bo1;
            out[(size_t)(r0 + j) * 3 + 2] = po[j][2] + bo2;
        }
    }
}

// ---------------------------------------------------------------------------
extern "C" void kernel_launch(void* const* d_in, const int* in_sizes, int n_in,
                              void* d_out, int out_size, void* d_ws, size_t ws_size,
                              hipStream_t stream) {
    (void)in_sizes; (void)n_in; (void)out_size; (void)ws_size;
    const float* ycoord = (const float*)d_in[0];
    const float* rndata = (const float*)d_in[1];
    const float* qcoord = (const float*)d_in[2];
    const float* w_e0 = (const float*)d_in[3];
    const float* b_e0 = (const float*)d_in[4];
    const float* w_e1 = (const float*)d_in[5];
    const float* b_e1 = (const float*)d_in[6];
    const float* w_e2 = (const float*)d_in[7];
    const float* b_e2 = (const float*)d_in[8];
    const float* w_sw0 = (const float*)d_in[9];
    const float* b_sw0 = (const float*)d_in[10];
    const float* w_sw1 = (const float*)d_in[11];
    const float* b_sw1 = (const float*)d_in[12];
    const float* w_p0 = (const float*)d_in[13];
    const float* b_p0 = (const float*)d_in[14];
    const float* w_p1 = (const float*)d_in[15];
    const float* b_p1 = (const float*)d_in[16];
    float* out = (float*)d_out;

    char* ws = (char*)d_ws;
    size_t off = 0;
    int*   topIdx  = (int*)(ws + off);   off += (size_t)TOTQ * KNN * sizeof(int);
    float* topD2   = (float*)(ws + off); off += (size_t)TOTQ * KNN * sizeof(float);
    float* decoded = (float*)(ws + off); off += (size_t)TOTQ * CINCH * sizeof(float);
    _Float16* fB1  = (_Float16*)(ws + off); off += (size_t)HDIM * HDIM * sizeof(_Float16);
    _Float16* fB2  = (_Float16*)(ws + off); off += (size_t)HDIM * HDIM * sizeof(_Float16);
    _Float16* fP0  = (_Float16*)(ws + off); off += (size_t)CINCH * PHDIM * sizeof(_Float16);

    prep_kernel<<<dim3(48), dim3(256), 0, stream>>>(w_e1, w_e2, w_p0, fB1, fB2, fP0);
    topk_kernel<<<dim3(TOTQ / TKWAVES), dim3(128), 0, stream>>>(ycoord, qcoord, topIdx, topD2);
    edge_kernel<<<dim3(TOTQ / EWAVES), dim3(256), 0, stream>>>(
        ycoord, rndata, qcoord, w_e0, b_e0, b_e1, b_e2,
        w_sw0, b_sw0, w_sw1, b_sw1, fB1, fB2, topIdx, topD2, decoded);
    proj_kernel<<<dim3(TOTQ / 16 / PWAVES), dim3(256), 0, stream>>>(
        decoded, fP0, b_p0, w_p1, b_p1, out);
}
